// CEABlock_ori_27247272526357
// MI455X (gfx1250) — matrix operand order repaired and checked by reading, not compiled, not run on hardware
//
#include <hip/hip_runtime.h>
#include <hip/hip_bf16.h>
#include <math.h>

// ---------------------------------------------------------------------------
// CDNA5 / gfx1250 types for WMMA
// ---------------------------------------------------------------------------
typedef __attribute__((ext_vector_type(16))) _Float16 v16h;
typedef __attribute__((ext_vector_type(8)))  _Float16 v8h;
typedef __attribute__((ext_vector_type(8)))  float    v8f;
typedef __attribute__((ext_vector_type(4)))  float    v4f_;

#define F_BIAS 1
#define F_GELU 2
#define F_RES  4

namespace {
constexpr int NB  = 32;            // batch
constexpr int LT  = 64;            // text tokens
constexpr int LS  = 256;           // spatial tokens
constexpr int LL  = LT + LS;       // 320
constexpr int CC  = 768;
constexpr int NH  = 12;
constexpr int HD  = CC / NH;       // 64
constexpr int LK  = 180;           // ceil(0.7*256)
constexpr int RREST = LS - LK;     // 76
constexpr int LP  = LT + LK;       // 244 tokens after pruning
constexpr int MLP = 4 * CC;        // 3072
constexpr float LNEPS = 1e-5f;
}

// ---------------------------------------------------------------------------
// fp32 -> f16 transpose+convert for weights: Wt[n][k] = (f16)W[k][n]
// LDS-tiled 32x32 so both global access directions are coalesced.
// grid = (N/32, K/32), block = 256 (32x8)
// ---------------------------------------------------------------------------
__global__ __launch_bounds__(256) void cvtT_f32_f16(const float* __restrict__ w,
                                                    _Float16* __restrict__ wt,
                                                    int K, int N) {
    __shared__ float tile[32][33];
    const int nt = blockIdx.x * 32, kt = blockIdx.y * 32;
    const int tx = threadIdx.x & 31, ty = threadIdx.x >> 5;
#pragma unroll
    for (int i = 0; i < 32; i += 8)
        tile[ty + i][tx] = w[(long long)(kt + ty + i) * N + nt + tx];
    __syncthreads();
#pragma unroll
    for (int i = 0; i < 32; i += 8)
        wt[(long long)(nt + ty + i) * K + kt + tx] = (_Float16)tile[tx][ty + i];
}

// ---------------------------------------------------------------------------
// Vt[b,h][d][m] = qkv[b*LL+m][2*CC + h*HD + d]  (f16 -> f16 transpose)
// grid = NB*NH blocks, each transposes its 320x64 V slab via 32x32 LDS tiles
// ---------------------------------------------------------------------------
__global__ __launch_bounds__(256) void transpose_v(const _Float16* __restrict__ qkv,
                                                   _Float16* __restrict__ vt) {
    __shared__ _Float16 tile[32][33];
    const int bh = blockIdx.x;
    const int b = bh / NH, h = bh % NH;
    const _Float16* src = qkv + (long long)b * LL * 3 * CC + 2 * CC + h * HD;
    _Float16* dst = vt + (long long)bh * HD * LL;
    const int tx = threadIdx.x & 31, ty = threadIdx.x >> 5;
    for (int mt = 0; mt < LL; mt += 32) {
        for (int dt = 0; dt < HD; dt += 32) {
#pragma unroll
            for (int i = 0; i < 32; i += 8)
                tile[ty + i][tx] = src[(long long)(mt + ty + i) * (3 * CC) + dt + tx];
            __syncthreads();
#pragma unroll
            for (int i = 0; i < 32; i += 8)
                dst[(long long)(dt + ty + i) * LL + mt + tx] = tile[tx][ty + i];
            __syncthreads();
        }
    }
}

__global__ void copy_i32(const int* __restrict__ s, int* __restrict__ d, int n) {
    int i = blockIdx.x * blockDim.x + threadIdx.x;
    if (i < n) d[i] = s[i];
}

// ---------------------------------------------------------------------------
// LayerNorm over C=768, one block (256 threads) per token, f16 output
// ---------------------------------------------------------------------------
__global__ __launch_bounds__(256) void ln_f16(const float* __restrict__ x,
                                              const float* __restrict__ w,
                                              const float* __restrict__ b,
                                              _Float16* __restrict__ o) {
    __shared__ float red[256];
    const int t = threadIdx.x;
    const long long row = blockIdx.x;
    const float* xr = x + row * CC;
    float v0 = xr[t], v1 = xr[t + 256], v2 = xr[t + 512];
    red[t] = v0 + v1 + v2;
    __syncthreads();
    for (int st = 128; st > 0; st >>= 1) { if (t < st) red[t] += red[t + st]; __syncthreads(); }
    const float mean = red[0] * (1.0f / CC);
    __syncthreads();
    const float d0 = v0 - mean, d1 = v1 - mean, d2 = v2 - mean;
    red[t] = d0 * d0 + d1 * d1 + d2 * d2;
    __syncthreads();
    for (int st = 128; st > 0; st >>= 1) { if (t < st) red[t] += red[t + st]; __syncthreads(); }
    const float rstd = rsqrtf(red[0] * (1.0f / CC) + LNEPS);
    _Float16* orow = o + row * CC;
    orow[t]       = (_Float16)(d0 * rstd * w[t]       + b[t]);
    orow[t + 256] = (_Float16)(d1 * rstd * w[t + 256] + b[t + 256]);
    orow[t + 512] = (_Float16)(d2 * rstd * w[t + 512] + b[t + 512]);
}

// ---------------------------------------------------------------------------
// Batched WMMA GEMM, zero LDS: both A and B fragments are loaded directly
// from global memory as b128 chunks in the exact ISA lane layout.
//   A: row-major, k-contiguous per lane (two 16B runs per fragment)
//   B: n-major / k-contiguous (i.e. B^T row-major): two 16B runs per fragment
// Block = 128 threads = 4 waves (2x2), block tile 64(M) x 64(N),
// each wave owns 32x32 = four 16x16 accumulators -> 4 WMMAs per 32-K slab.
// Cross-wave/block reuse of A and B is served by the 192MB L2.
// A-element type is a template parameter so the inner loop is branch-free.
// ---------------------------------------------------------------------------
struct GemmP {
    const void*      A;     // _Float16* or float* (AF32)
    const _Float16*  Bm;    // n-major, k-contiguous
    float*           Cf;    // optional fp32 out
    _Float16*        Ch;    // optional f16 out
    const float*     bias;  // [N]
    const float*     res;   // optional residual, same geometry as C
    long long a_sb, a_sh, b_sb, b_sh, c_sb, c_sh, r_sb, r_sh;
    int lda, ldb, ldc;
    int M, N, K, nH, flags, a_f32;
    float scale;
};

__device__ __forceinline__ v16h frag_cat(v8h lo, v8h hi) {
    return __builtin_shufflevector(lo, hi, 0, 1, 2, 3, 4, 5, 6, 7,
                                   8, 9, 10, 11, 12, 13, 14, 15);
}

template <bool AF32>
__global__ __launch_bounds__(128) void gemm_wmma_f16(GemmP p) {
    const int tid = threadIdx.x;
    const int bx = blockIdx.x, by = blockIdx.y, bz = blockIdx.z;
    const int nH = (p.nH > 0) ? p.nH : 1;
    const int bb = bz / nH, hh = bz % nH;

    const long long aoff = (long long)bb * p.a_sb + (long long)hh * p.a_sh;
    const long long coff = (long long)bb * p.c_sb + (long long)hh * p.c_sh;
    const long long roff = (long long)bb * p.r_sb + (long long)hh * p.r_sh;

    const int m0 = by * 64, n0 = bx * 64;
    const int wave = tid >> 5, lane = tid & 31;
    const int wm = wave & 1, wn = wave >> 1;     // 2x2 wave grid
    const int lr = lane & 15, lhi = lane >> 4;

    // ISA 16-bit A layout: per lane two contiguous 16B runs (K kb..kb+7, kb+16..kb+23)
    const int arow0 = m0 + wm * 32 + lr;
    const int akb = lhi * 8;
    // ISA 16-bit B layout: per lane two contiguous 16B runs (K kb..kb+7, kb+16..kb+23)
    const int bcol0 = n0 + wn * 32 + lr;
    const int bkb = lhi * 8;

    // Loop-invariant base pointers, bumped by +32 elements per K slab.
    const float*    paf = (const float*)p.A + aoff + (long long)arow0 * p.lda + akb;
    const _Float16* pah = (const _Float16*)p.A + aoff + (long long)arow0 * p.lda + akb;
    const _Float16* pb  = p.Bm + (long long)bb * p.b_sb + (long long)hh * p.b_sh
                        + (long long)bcol0 * p.ldb + bkb;
    const long long aRow16 = (long long)16 * p.lda;
    const long long bCol16 = (long long)16 * p.ldb;

    v8f acc00 = {}, acc01 = {}, acc10 = {}, acc11 = {};

    for (int k0 = 0; k0 < p.K; k0 += 32) {
        v16h a0, a1;
        if constexpr (AF32) {
            v4f_ f00 = *(const v4f_*)(paf);
            v4f_ f01 = *(const v4f_*)(paf + 4);
            v4f_ f02 = *(const v4f_*)(paf + 16);
            v4f_ f03 = *(const v4f_*)(paf + 20);
            v4f_ f10 = *(const v4f_*)(paf + aRow16);
            v4f_ f11 = *(const v4f_*)(paf + aRow16 + 4);
            v4f_ f12 = *(const v4f_*)(paf + aRow16 + 16);
            v4f_ f13 = *(const v4f_*)(paf + aRow16 + 20);
#pragma unroll
            for (int e = 0; e < 4; ++e) {
                a0[e] = (_Float16)f00[e]; a0[e + 4]  = (_Float16)f01[e];
                a0[e + 8] = (_Float16)f02[e]; a0[e + 12] = (_Float16)f03[e];
                a1[e] = (_Float16)f10[e]; a1[e + 4]  = (_Float16)f11[e];
                a1[e + 8] = (_Float16)f12[e]; a1[e + 12] = (_Float16)f13[e];
            }
            paf += 32;
            __builtin_prefetch(paf, 0, 1);           // speculative, next K slab
        } else {
            a0 = frag_cat(*(const v8h*)pah, *(const v8h*)(pah + 16));
            a1 = frag_cat(*(const v8h*)(pah + aRow16), *(const v8h*)(pah + aRow16 + 16));
            pah += 32;
            __builtin_prefetch(pah, 0, 1);           // speculative, next K slab
        }

        v16h b0 = frag_cat(*(const v8h*)pb, *(const v8h*)(pb + 16));
        v16h b1 = frag_cat(*(const v8h*)(pb + bCol16), *(const v8h*)(pb + bCol16 + 16));
        pb += 32;
        __builtin_prefetch(pb, 0, 1);                // speculative, next K slab

        acc00 = __builtin_amdgcn_wmma_f32_16x16x32_f16(false, a0, false, b0, (short)0, acc00, false, false);
        acc01 = __builtin_amdgcn_wmma_f32_16x16x32_f16(false, a0, false, b1, (short)0, acc01, false, false);
        acc10 = __builtin_amdgcn_wmma_f32_16x16x32_f16(false, a1, false, b0, (short)0, acc10, false, false);
        acc11 = __builtin_amdgcn_wmma_f32_16x16x32_f16(false, a1, false, b1, (short)0, acc11, false, false);
    }

    // ---- epilogue: C/D layout VGPR r -> M = r + (lane>=16)*8, N = lane&15 ----
    v8f accs[2][2] = { { acc00, acc01 }, { acc10, acc11 } };
    const int mrow = m0 + wm * 32 + lhi * 8;
    const int ncol = n0 + wn * 32 + lr;
#pragma unroll
    for (int mt = 0; mt < 2; ++mt) {
#pragma unroll
        for (int nt = 0; nt < 2; ++nt) {
            const int nc = ncol + nt * 16;
            const float bv = (p.flags & F_BIAS) ? p.bias[nc] : 0.0f;
#pragma unroll
            for (int r = 0; r < 8; ++r) {
                const int m = mrow + mt * 16 + r;
                float v = accs[mt][nt][r] * p.scale + bv;
                if (p.flags & F_GELU) {
                    const float u = v;
                    v = 0.5f * u * (1.0f + tanhf(0.7978845608028654f * (u + 0.044715f * u * u * u)));
                }
                if (p.flags & F_RES)
                    v += p.res[roff + (long long)m * p.ldc + nc];
                const long long ci = coff + (long long)m * p.ldc + nc;
                if (p.Cf) p.Cf[ci] = v;
                if (p.Ch) p.Ch[ci] = (_Float16)v;
            }
        }
    }
}

// ---------------------------------------------------------------------------
// Row softmax in place over L=320 columns (attn lives in d_out, fp32)
// ---------------------------------------------------------------------------
__global__ __launch_bounds__(256) void softmax_inplace(float* __restrict__ attn) {
    __shared__ float red[256];
    const long long row = blockIdx.x;
    float* p = attn + row * (long long)LL;
    const int t = threadIdx.x;
    const float a0 = p[t];
    const float a1 = (t < LL - 256) ? p[t + 256] : -3.4e38f;
    red[t] = fmaxf(a0, a1);
    __syncthreads();
    for (int st = 128; st > 0; st >>= 1) { if (t < st) red[t] = fmaxf(red[t], red[t + st]); __syncthreads(); }
    const float mx = red[0];
    __syncthreads();
    const float e0 = __expf(a0 - mx);
    const float e1 = (t < LL - 256) ? __expf(a1 - mx) : 0.0f;
    red[t] = e0 + e1;
    __syncthreads();
    for (int st = 128; st > 0; st >>= 1) { if (t < st) red[t] += red[t + st]; __syncthreads(); }
    const float inv = 1.0f / red[0];
    p[t] = e0 * inv;
    if (t < LL - 256) p[t + 256] = e1 * inv;
}

// ---------------------------------------------------------------------------
// attn_t[b][j] = mean over (h, i<LT) of attn[b,h,i,LT+j]   (j in 0..LS-1)
// ---------------------------------------------------------------------------
__global__ __launch_bounds__(256) void meanpool(const float* __restrict__ attn,
                                                float* __restrict__ attn_t) {
    const int b = blockIdx.x, j = threadIdx.x;
    const float* base = attn + (long long)b * NH * LL * LL;
    float s = 0.0f;
    for (int h = 0; h < NH; ++h) {
        const float* ph = base + (long long)h * LL * LL + LT + j;
        for (int i = 0; i < LT; ++i) s += ph[(long long)i * LL];
    }
    attn_t[b * LS + j] = s * (1.0f / (NH * LT));
}

// ---------------------------------------------------------------------------
// Per-batch bitonic sort of 256 (value, index) pairs, descending value,
// stable tie-break on ascending index (== jnp.argsort(-x) semantics).
// ---------------------------------------------------------------------------
__global__ __launch_bounds__(256) void topk_sort(const float* __restrict__ attn_t,
                                                 const int* __restrict__ gi,
                                                 int* __restrict__ keep,
                                                 int* __restrict__ removed,
                                                 int* __restrict__ sortidx) {
    __shared__ float sv[256];
    __shared__ int   si[256];
    const int b = blockIdx.x, t = threadIdx.x;
    sv[t] = attn_t[b * LS + t];
    si[t] = t;
    __syncthreads();
    for (int k = 2; k <= 256; k <<= 1) {
        for (int j = k >> 1; j > 0; j >>= 1) {
            const int x = t ^ j;
            if (x > t) {
                const bool up = ((t & k) == 0);
                const float av = sv[t]; const int ai = si[t];
                const float bvv = sv[x]; const int bi = si[x];
                const bool lhsFirst = (av > bvv) || (av == bvv && ai < bi);
                if (lhsFirst != up) { sv[t] = bvv; si[t] = bi; sv[x] = av; si[x] = ai; }
            }
            __syncthreads();
        }
    }
    const int idx = si[t];
    sortidx[b * LS + t] = idx;
    if (t < LK) keep[b * LK + t] = gi[b * LS + idx];
    else        removed[b * RREST + (t - LK)] = gi[b * LS + idx];
}

// ---------------------------------------------------------------------------
// Assemble pruned token sequence: [0..LT) passthrough, [LT..LP) gathered
// ---------------------------------------------------------------------------
__global__ __launch_bounds__(256) void gather_tokens(const float* __restrict__ xres,
                                                     const int* __restrict__ sortidx,
                                                     float* __restrict__ xo) {
    const int row = blockIdx.x;
    const int b = row / LP, l = row % LP;
    const int src = (l < LT) ? l : (LT + sortidx[b * LS + (l - LT)]);
    const float* s = xres + ((long long)b * LL + src) * CC;
    float* d = xo + (long long)row * CC;
    const int t = threadIdx.x;
    d[t] = s[t];
    d[t + 256] = s[t + 256];
    d[t + 512] = s[t + 512];
}

// ---------------------------------------------------------------------------
// Host orchestration
// ---------------------------------------------------------------------------
static inline void launch_gemm(const GemmP& p, int batches, hipStream_t st) {
    dim3 g(p.N / 64, p.M / 64, batches);
    if (p.a_f32) gemm_wmma_f16<true><<<g, 128, 0, st>>>(p);
    else         gemm_wmma_f16<false><<<g, 128, 0, st>>>(p);
}

extern "C" void kernel_launch(void* const* d_in, const int* in_sizes, int n_in,
                              void* d_out, int out_size, void* d_ws, size_t ws_size,
                              hipStream_t stream) {
    (void)in_sizes; (void)n_in; (void)out_size; (void)ws_size;
    const float* xin[2]  = { (const float*)d_in[0], (const float*)d_in[1] };
    const int* gi_t[2]   = { (const int*)d_in[2],   (const int*)d_in[3] };
    const int* gi_s[2]   = { (const int*)d_in[4],   (const int*)d_in[5] };
    const float* n1w = (const float*)d_in[6],  *n1b = (const float*)d_in[7];
    const float* qkvw = (const float*)d_in[8], *qkvb = (const float*)d_in[9];
    const float* projw = (const float*)d_in[10], *projb = (const float*)d_in[11];
    const float* n2w = (const float*)d_in[12], *n2b = (const float*)d_in[13];
    const float* fc1w = (const float*)d_in[14], *fc1b = (const float*)d_in[15];
    const float* fc2w = (const float*)d_in[16], *fc2b = (const float*)d_in[17];
    float* outf = (float*)d_out;

    // ---- output layout (flat, return order) ----
    const long long SZ_X = (long long)NB * LP * CC;
    const long long SZ_ATTN = (long long)NB * NH * LL * LL;
    struct Outs { float* x; int* git; int* keep; int* rem; float* attn; } O[2];
    long long off = 0;
    for (int s = 0; s < 2; ++s) {
        O[s].x    = outf + off;           off += SZ_X;
        O[s].git  = (int*)(outf + off);   off += (long long)NB * LT;
        O[s].keep = (int*)(outf + off);   off += (long long)NB * LK;
        O[s].rem  = (int*)(outf + off);   off += (long long)NB * RREST;
        O[s].attn = outf + off;           off += SZ_ATTN;
    }

    // ---- workspace layout ----
    char* wsp = (char*)d_ws;
    auto walloc = [&](size_t bytes) -> char* {
        char* p = wsp;
        wsp += (bytes + 255) & ~(size_t)255;
        return p;
    };
    _Float16* qkvw_t  = (_Float16*)walloc((size_t)CC * 3 * CC * 2);   // [2304][768]
    _Float16* projw_t = (_Float16*)walloc((size_t)CC * CC * 2);       // [768][768]
    _Float16* fc1w_t  = (_Float16*)walloc((size_t)CC * MLP * 2);      // [3072][768]
    _Float16* fc2w_t  = (_Float16*)walloc((size_t)MLP * CC * 2);      // [768][3072]
    _Float16* ln1buf  = (_Float16*)walloc((size_t)NB * LL * CC * 2);
    _Float16* qkvbuf  = (_Float16*)walloc((size_t)NB * LL * 3 * CC * 2);
    _Float16* vtbuf   = (_Float16*)walloc((size_t)NB * NH * HD * LL * 2);
    _Float16* avout_h = (_Float16*)walloc((size_t)NB * LL * CC * 2);
    float*    xres    = (float*)walloc((size_t)NB * LL * CC * 4);
    float*    attn_t  = (float*)walloc((size_t)NB * LS * 4);
    int*      sortidx = (int*)walloc((size_t)NB * LS * 4);
    _Float16* ln2buf  = (_Float16*)walloc((size_t)NB * LP * CC * 2);
    _Float16* hbuf    = (_Float16*)walloc((size_t)NB * LP * MLP * 2);

    // ---- weights: transpose+convert to f16 (k-contiguous B layout) ----
    auto cvtT = [&](const float* src, _Float16* dst, int K, int N) {
        dim3 g(N / 32, K / 32);
        cvtT_f32_f16<<<g, 256, 0, stream>>>(src, dst, K, N);
    };
    cvtT(qkvw,  qkvw_t,  CC, 3 * CC);
    cvtT(projw, projw_t, CC, CC);
    cvtT(fc1w,  fc1w_t,  CC, MLP);
    cvtT(fc2w,  fc2w_t,  MLP, CC);

    for (int s = 0; s < 2; ++s) {
        // gi_t passthrough
        copy_i32<<<(NB * LT + 255) / 256, 256, 0, stream>>>(gi_t[s], O[s].git, NB * LT);

        // 1) LN1 -> f16
        ln_f16<<<NB * LL, 256, 0, stream>>>(xin[s], n1w, n1b, ln1buf);

        // 2) QKV GEMM: (10240 x 768) @ (768 x 2304), f16 out
        {
            GemmP p{};
            p.A = ln1buf; p.Bm = qkvw_t; p.Ch = qkvbuf; p.bias = qkvb;
            p.lda = CC; p.ldb = CC; p.ldc = 3 * CC;
            p.M = NB * LL; p.N = 3 * CC; p.K = CC; p.nH = 1;
            p.flags = F_BIAS; p.scale = 1.0f;
            launch_gemm(p, 1, stream);
        }

        // 2b) V^T slabs for the attn@V GEMM
        transpose_v<<<NB * NH, 256, 0, stream>>>(qkvbuf, vtbuf);

        // 3) Scores: per (b,h): q(320x64) @ k^T(64x320) * 1/8 -> attn (fp32).
        //    K rows are d-fastest in qkvbuf, i.e. already k-contiguous B.
        {
            GemmP p{};
            p.A = qkvbuf; p.a_sb = (long long)LL * 3 * CC; p.a_sh = HD; p.lda = 3 * CC;
            p.Bm = qkvbuf + CC; p.b_sb = (long long)LL * 3 * CC; p.b_sh = HD; p.ldb = 3 * CC;
            p.Cf = O[s].attn; p.c_sb = (long long)NH * LL * LL; p.c_sh = (long long)LL * LL; p.ldc = LL;
            p.M = LL; p.N = LL; p.K = HD; p.nH = NH;
            p.scale = 0.125f;   // HD^-0.5
            launch_gemm(p, NB * NH, stream);
        }

        // 4) softmax rows, in place (this IS the returned attn tensor)
        softmax_inplace<<<NB * NH * LL, 256, 0, stream>>>(O[s].attn);

        // 5) attn @ V: A fp32 (converted in regs), B = V^T (k-contiguous)
        {
            GemmP p{};
            p.A = O[s].attn; p.a_f32 = 1;
            p.a_sb = (long long)NH * LL * LL; p.a_sh = (long long)LL * LL; p.lda = LL;
            p.Bm = vtbuf; p.b_sb = (long long)NH * HD * LL; p.b_sh = (long long)HD * LL; p.ldb = LL;
            p.Ch = avout_h; p.c_sb = (long long)LL * CC; p.c_sh = HD; p.ldc = CC;
            p.M = LL; p.N = HD; p.K = LL; p.nH = NH; p.scale = 1.0f;
            launch_gemm(p, NB * NH, stream);
        }

        // 6) proj GEMM + bias + residual(x) -> xres fp32
        {
            GemmP p{};
            p.A = avout_h; p.Bm = projw_t; p.Cf = xres; p.bias = projb; p.res = xin[s];
            p.lda = CC; p.ldb = CC; p.ldc = CC;
            p.M = NB * LL; p.N = CC; p.K = CC; p.nH = 1;
            p.flags = F_BIAS | F_RES; p.scale = 1.0f;
            launch_gemm(p, 1, stream);
        }

        // 7) token scoring + stable top-k + gather into d_out x region
        meanpool<<<NB, 256, 0, stream>>>(O[s].attn, attn_t);
        topk_sort<<<NB, 256, 0, stream>>>(attn_t, gi_s[s], O[s].keep, O[s].rem, sortidx);
        gather_tokens<<<NB * LP, 256, 0, stream>>>(xres, sortidx, O[s].x);

        // 8) MLP: LN2 -> fc1 (+GELU) -> fc2 (+residual into d_out)
        ln_f16<<<NB * LP, 256, 0, stream>>>(O[s].x, n2w, n2b, ln2buf);
        {
            GemmP p{};
            p.A = ln2buf; p.Bm = fc1w_t; p.Ch = hbuf; p.bias = fc1b;
            p.lda = CC; p.ldb = CC; p.ldc = MLP;
            p.M = NB * LP; p.N = MLP; p.K = CC; p.nH = 1;
            p.flags = F_BIAS | F_GELU; p.scale = 1.0f;
            launch_gemm(p, 1, stream);
        }
        {
            GemmP p{};
            p.A = hbuf; p.Bm = fc2w_t; p.Cf = O[s].x; p.bias = fc2b; p.res = O[s].x;
            p.lda = MLP; p.ldb = MLP; p.ldc = CC;
            p.M = NB * LP; p.N = CC; p.K = MLP; p.nH = 1;
            p.flags = F_BIAS | F_RES; p.scale = 1.0f;
            launch_gemm(p, 1, stream);
        }
    }
}
